// CrossGate_57801669870305
// MI455X (gfx1250) — compile-verified, matrix operand order checked by reading, not run on hardware
//
#include <hip/hip_runtime.h>
#include <math.h>

// ---------------- types ----------------
typedef __attribute__((ext_vector_type(16))) _Float16 h16;
typedef __attribute__((ext_vector_type(8)))  _Float16 h8;
typedef __attribute__((ext_vector_type(8)))  float    f8;

union H16 { h16 v; h8 h[2]; };

// ---------------- optional gfx1250 async global->LDS path ----------------
#if defined(__AMDGCN__) && __has_builtin(__builtin_amdgcn_global_load_async_to_lds_b128)
#define HAVE_ASYNC_LDS 1
#define AS1 __attribute__((address_space(1)))
#define AS3 __attribute__((address_space(3)))
typedef int v4i_ __attribute__((vector_size(16)));
__device__ __forceinline__ void async_b128(const _Float16* g, _Float16* l) {
  __builtin_amdgcn_global_load_async_to_lds_b128(
      (AS1 v4i_*)(unsigned long long)(g),
      (AS3 v4i_*)(unsigned)(unsigned long long)(l), 0, 0);
}
__device__ __forceinline__ void wait_async0() {
#if __has_builtin(__builtin_amdgcn_s_wait_asynccnt)
  __builtin_amdgcn_s_wait_asynccnt(0);
#else
  asm volatile("s_wait_asynccnt 0" ::: "memory");
#endif
}
#endif

// ---------------- static dims ----------------
constexpr int NB   = 4;     // batch
constexpr int NSP  = 576;   // H*W
constexpr int BNR  = 2304;  // B*N rows
constexpr int CIN  = 128;   // x/y channels
constexpr int QKC  = 512;   // num_head*dim_k
constexpr int DHC  = 2048;  // num_head*d_v
constexpr int NHD  = 8;
constexpr int DKH  = 64;
constexpr int DVH  = 256;
constexpr int A3C  = 192;   // 3*ADIM
constexpr int ADC  = 64;    // ADIM
constexpr int HIDC = 512;   // uffn hidden

__device__ __forceinline__ float act_apply(float x, int act) {
  if (act == 1) return x > 0.f ? x : 0.f;
  if (act == 2) return 0.5f * x * (1.f + erff(x * 0.70710678118654752f));
  return x;
}

// load one 32-half row chunk from global (guarded, zero-padded)
__device__ __forceinline__ void load_row32(const _Float16* __restrict__ P,
                                           int row, int rows, int k0, int K, int ld,
                                           H16& r0, H16& r1)
{
  if (row < rows && (k0 + 32) <= K) {
    const _Float16* p = P + (size_t)row * ld + k0;
    r0.h[0] = *(const h8*)(p);      r0.h[1] = *(const h8*)(p + 8);
    r1.h[0] = *(const h8*)(p + 16); r1.h[1] = *(const h8*)(p + 24);
  } else {
#pragma unroll
    for (int i = 0; i < 16; ++i) {
      r0.v[i] = (row < rows && (k0 + i)      < K) ? P[(size_t)row * ld + k0 + i]      : (_Float16)0.f;
      r1.v[i] = (row < rows && (k0 + 16 + i) < K) ? P[(size_t)row * ld + k0 + 16 + i] : (_Float16)0.f;
    }
  }
}

__device__ __forceinline__ void store_row32(_Float16* d, const H16& r0, const H16& r1)
{
  *(h8*)(d)      = r0.h[0]; *(h8*)(d + 8)  = r0.h[1];
  *(h8*)(d + 16) = r1.h[0]; *(h8*)(d + 24) = r1.h[1];
}

// =====================================================================
// Generic fused WMMA GEMM, double-buffered LDS pipeline (+async copies).
//   For z = blockIdx.z (z = zb*ZH + zh):
//   Out[n*ldo + m] = act((sum_k A[m*lda+k]*B[n*ldb+k]) * scale[m] + shift[m]) + res
//   A rows indexed by m (M of them), B rows indexed by n (N of them).
//   WMMA M-dim = m (memory-inner axis of Out) so D stores are contiguous h8.
// =====================================================================
__global__ __launch_bounds__(128)
void k_wmma_gemm(const _Float16* __restrict__ Aptr,
                 const _Float16* __restrict__ Bptr,
                 _Float16* __restrict__ Optr,
                 int M, int N, int K,
                 int lda, int ldb, int ldo,
                 int ZH, int aZb, int aZh, int bZb, int bZh, int oZb, int oZh,
                 const float* __restrict__ sv, const float* __restrict__ bv,
                 float kscale, int act,
                 const _Float16* __restrict__ res)
{
  const int z  = blockIdx.z;
  const int zb = z / ZH, zh = z - zb * ZH;
  Aptr += (size_t)zb * aZb + (size_t)zh * aZh;
  Bptr += (size_t)zb * bZb + (size_t)zh * bZh;
  Optr += (size_t)zb * oZb + (size_t)zh * oZh;
  const _Float16* Rptr = res ? (res + (size_t)zb * oZb + (size_t)zh * oZh) : nullptr;

  const int m_blk = blockIdx.x * 128;
  const int n_blk = blockIdx.y * 128;
  const int tid   = threadIdx.x;
  const int lane  = tid & 31;
  const int wave  = tid >> 5;
  const int wm    = (wave & 1) * 64;
  const int wn    = (wave >> 1) * 64;
  const bool wok  = (m_blk + wm < M) && (n_blk + wn < N);
  const int l16   = lane & 15;
  const int kh    = (lane >> 4) * 8;   // wmma 16-bit frag K-offset per lane half

  __shared__ _Float16 sA[2][128 * 40];   // 32 K-halfs + 8 pad per row, double buffered
  __shared__ _Float16 sB[2][128 * 40];

  f8 acc[4][4];
#pragma unroll
  for (int i = 0; i < 4; ++i)
#pragma unroll
    for (int j = 0; j < 4; ++j)
#pragma unroll
      for (int r = 0; r < 8; ++r) acc[i][j][r] = 0.f;

  const int nk = (K + 31) >> 5;
  const int ar = m_blk + tid;
  const int br = n_blk + tid;
  const bool tileAfull = (m_blk + 128) <= M;   // uniform: whole A tile in range
  const bool tileBfull = (n_blk + 128) <= N;   // uniform: whole B tile in range

  H16 ra0, ra1, rb0, rb1;
  load_row32(Aptr, ar, M, 0, K, lda, ra0, ra1);
  load_row32(Bptr, br, N, 0, K, ldb, rb0, rb1);
  store_row32(&sA[0][tid * 40], ra0, ra1);
  store_row32(&sB[0][tid * 40], rb0, rb1);
  __syncthreads();

  for (int i = 0; i < nk; ++i) {
    const int cur   = i & 1;
    const bool more = (i + 1) < nk;
    bool regA = false, regB = false;   // regs hold next tile -> needs ds_store
    if (more) {      // issue next tile's fetches before compute (latency overlap)
      const int k1 = (i + 1) * 32;
      const bool kfull = (k1 + 32) <= K;
#if HAVE_ASYNC_LDS
      if (tileAfull && kfull) {        // async copy engine: global -> LDS, no VGPRs
        const _Float16* g = Aptr + (size_t)ar * lda + k1;
        _Float16* l = &sA[cur ^ 1][tid * 40];
        async_b128(g, l);           async_b128(g + 8,  l + 8);
        async_b128(g + 16, l + 16); async_b128(g + 24, l + 24);
      } else { load_row32(Aptr, ar, M, k1, K, lda, ra0, ra1); regA = true; }
      if (tileBfull && kfull) {
        const _Float16* g = Bptr + (size_t)br * ldb + k1;
        _Float16* l = &sB[cur ^ 1][tid * 40];
        async_b128(g, l);           async_b128(g + 8,  l + 8);
        async_b128(g + 16, l + 16); async_b128(g + 24, l + 24);
      } else { load_row32(Bptr, br, N, k1, K, ldb, rb0, rb1); regB = true; }
#else
      (void)tileAfull; (void)tileBfull; (void)kfull;
      load_row32(Aptr, ar, M, k1, K, lda, ra0, ra1); regA = true;
      load_row32(Bptr, br, N, k1, K, ldb, rb0, rb1); regB = true;
#endif
    }
    if (i + 2 < nk) {  // keep L2->WGP ahead: global_prefetch_b8
      const int k2 = (i + 2) * 32;
      if (ar < M) __builtin_prefetch(Aptr + (size_t)ar * lda + k2, 0, 0);
      if (br < N) __builtin_prefetch(Bptr + (size_t)br * ldb + k2, 0, 0);
    }
    if (wok) {
      H16 af[4], bf[4];
#pragma unroll
      for (int t = 0; t < 4; ++t) {
        const _Float16* pa = &sA[cur][(wm + t * 16 + l16) * 40 + kh];
        af[t].h[0] = *(const h8*)(pa);
        af[t].h[1] = *(const h8*)(pa + 16);
        const _Float16* pb = &sB[cur][(wn + t * 16 + l16) * 40 + kh];
        bf[t].h[0] = *(const h8*)(pb);
        bf[t].h[1] = *(const h8*)(pb + 16);
      }
#pragma unroll
      for (int ti = 0; ti < 4; ++ti)
#pragma unroll
        for (int tj = 0; tj < 4; ++tj)
          acc[ti][tj] = __builtin_amdgcn_wmma_f32_16x16x32_f16(
              false, af[ti].v, false, bf[tj].v, (short)0, acc[ti][tj], false, false);
    }
    if (more) {
      if (regA) store_row32(&sA[cur ^ 1][tid * 40], ra0, ra1);
      if (regB) store_row32(&sB[cur ^ 1][tid * 40], rb0, rb1);
    }
#if HAVE_ASYNC_LDS
    wait_async0();                   // async copies visible in LDS before barrier
#endif
    __syncthreads();
  }

  if (!wok) return;
  const int mo = (lane >> 4) * 8;
#pragma unroll
  for (int j = 0; j < 4; ++j) {
    const int n = n_blk + wn + j * 16 + l16;
#pragma unroll
    for (int i = 0; i < 4; ++i) {
      const int m0 = m_blk + wm + i * 16 + mo;
      f8 v = acc[i][j];
      if (sv) {
        const f8 s8 = *(const f8*)(sv + m0);
        const f8 b8 = *(const f8*)(bv + m0);
#pragma unroll
        for (int r = 0; r < 8; ++r) v[r] = v[r] * s8[r] + b8[r];
      } else if (kscale != 1.f) {
#pragma unroll
        for (int r = 0; r < 8; ++r) v[r] *= kscale;
      }
      if (act) {
#pragma unroll
        for (int r = 0; r < 8; ++r) v[r] = act_apply(v[r], act);
      }
      if (Rptr) {
        const h8 rv = *(const h8*)(Rptr + (size_t)n * ldo + m0);
#pragma unroll
        for (int r = 0; r < 8; ++r) v[r] += (float)rv[r];
      }
      h8 o;
#pragma unroll
      for (int r = 0; r < 8; ++r) o[r] = (_Float16)v[r];
      *(h8*)(Optr + (size_t)n * ldo + m0) = o;
    }
  }
}

// =====================================================================
// Talking-heads + softmax over att[b][n][h][m] (in place).
// One wave32 per (b,n); 8x576 mixed logits staged in LDS.
// useTH=0 -> plain per-head softmax (cross attention).
// =====================================================================
__global__ __launch_bounds__(32)
void k_softmax_th(_Float16* __restrict__ att,
                  const float* __restrict__ w1, const float* __restrict__ b1,
                  const float* __restrict__ w2, const float* __restrict__ b2,
                  int useTH)
{
  _Float16* base = att + (size_t)blockIdx.x * (NHD * NSP);
  const int lane = threadIdx.x;

  __shared__ float s[NHD][NSP];
  __shared__ float sw1[64], sb1[8], sw2[64], sb2[8];
  if (useTH) {
    if (lane < 8) { sb1[lane] = b1[lane]; sb2[lane] = b2[lane]; }
    for (int i = lane; i < 64; i += 32) { sw1[i] = w1[i]; sw2[i] = w2[i]; }
  }
  __syncthreads();

  for (int m = lane; m < NSP; m += 32) {
    float a[8];
#pragma unroll
    for (int h = 0; h < 8; ++h) a[h] = (float)base[h * NSP + m];
    if (useTH) {
#pragma unroll
      for (int g = 0; g < 8; ++g) {
        float t = sb1[g];
#pragma unroll
        for (int h = 0; h < 8; ++h) t += sw1[g * 8 + h] * a[h];
        s[g][m] = t;
      }
    } else {
#pragma unroll
      for (int h = 0; h < 8; ++h) s[h][m] = a[h];
    }
  }
  __syncthreads();

  float mx[8], sm[8];
#pragma unroll
  for (int g = 0; g < 8; ++g) { mx[g] = -3.0e38f; sm[g] = 0.f; }
  for (int m = lane; m < NSP; m += 32)
#pragma unroll
    for (int g = 0; g < 8; ++g) mx[g] = fmaxf(mx[g], s[g][m]);
#pragma unroll
  for (int g = 0; g < 8; ++g)
    for (int o = 16; o > 0; o >>= 1) mx[g] = fmaxf(mx[g], __shfl_xor(mx[g], o, 32));

  for (int m = lane; m < NSP; m += 32)
#pragma unroll
    for (int g = 0; g < 8; ++g) {
      float e = __expf(s[g][m] - mx[g]);
      s[g][m] = e;
      sm[g] += e;
    }
#pragma unroll
  for (int g = 0; g < 8; ++g)
    for (int o = 16; o > 0; o >>= 1) sm[g] += __shfl_xor(sm[g], o, 32);
  float inv[8];
#pragma unroll
  for (int g = 0; g < 8; ++g) inv[g] = 1.f / sm[g];
  __syncthreads();

  for (int m = lane; m < NSP; m += 32) {
    float p[8];
#pragma unroll
    for (int h = 0; h < 8; ++h) p[h] = s[h][m] * inv[h];
    if (useTH) {
#pragma unroll
      for (int g = 0; g < 8; ++g) {
        float t = sb2[g];
#pragma unroll
        for (int h = 0; h < 8; ++h) t += sw2[g * 8 + h] * p[h];
        base[g * NSP + m] = (_Float16)t;
      }
    } else {
#pragma unroll
      for (int h = 0; h < 8; ++h) base[h * NSP + m] = (_Float16)p[h];
    }
  }
}

// =====================================================================
// Cross-attention A@V, d=8 per head (tiny K -> VALU).
// att: [b][n][h][m] f16 ; qkvT: [b*N][192] (v slice at +128+h*8) ; out [b*N][64]
// =====================================================================
__global__ __launch_bounds__(256)
void k_cross_av(const _Float16* __restrict__ att,
                const _Float16* __restrict__ qkvT,
                _Float16* __restrict__ outT)
{
  const int id = blockIdx.x * 256 + threadIdx.x;     // (b*576+n)*8 + h
  if (id >= NB * NSP * NHD) return;
  const int h  = id & 7;
  const int bn = id >> 3;
  const int b  = bn / NSP;
  const _Float16* arow  = att + (size_t)id * NSP;
  const _Float16* vbase = qkvT + (size_t)b * NSP * A3C + 128 + h * 8;
  float acc[8];
#pragma unroll
  for (int d = 0; d < 8; ++d) acc[d] = 0.f;
  for (int m = 0; m < NSP; ++m) {
    const float a  = (float)arow[m];
    const h8 vr = *(const h8*)(vbase + (size_t)m * A3C);
#pragma unroll
    for (int d = 0; d < 8; ++d) acc[d] += a * (float)vr[d];
  }
  h8 o;
#pragma unroll
  for (int d = 0; d < 8; ++d) o[d] = (_Float16)acc[d];
  *(h8*)(outT + (size_t)bn * ADC + h * 8) = o;
}

// =====================================================================
// Depthwise 3x3, NCHW f16 in/out, fused BN scale/shift. thread = (b,c,h-row)
// =====================================================================
__global__ __launch_bounds__(256)
void k_dwconv(const _Float16* __restrict__ inN, const float* __restrict__ w9,
              const float* __restrict__ sv, const float* __restrict__ bv,
              _Float16* __restrict__ outN)
{
  const int id = blockIdx.x * 256 + threadIdx.x;
  if (id >= NB * DHC * 24) return;
  const int h  = id % 24;
  const int bc = id / 24;
  const int c  = bc % DHC;
  const _Float16* p = inN + (size_t)bc * NSP;

  float wv[9];
#pragma unroll
  for (int t = 0; t < 9; ++t) wv[t] = w9[c * 9 + t];
  const float sc = sv[c], sh = bv[c];

  float pad[3][26];
#pragma unroll
  for (int r = 0; r < 3; ++r) {
    pad[r][0] = 0.f; pad[r][25] = 0.f;
    const int hy = h - 1 + r;
    if (hy >= 0 && hy < 24) {
      const h8* rp = (const h8*)(p + hy * 24);
      const h8 v0 = rp[0], v1 = rp[1], v2 = rp[2];
#pragma unroll
      for (int x = 0; x < 8; ++x) {
        pad[r][1 + x]  = (float)v0[x];
        pad[r][9 + x]  = (float)v1[x];
        pad[r][17 + x] = (float)v2[x];
      }
    } else {
#pragma unroll
      for (int x = 0; x < 24; ++x) pad[r][1 + x] = 0.f;
    }
  }
  float o[24];
#pragma unroll
  for (int w = 0; w < 24; ++w) {
    float a = 0.f;
#pragma unroll
    for (int r = 0; r < 3; ++r)
#pragma unroll
      for (int t = 0; t < 3; ++t) a += wv[r * 3 + t] * pad[r][w + t];
    o[w] = a * sc + sh;
  }
  _Float16* op = outN + (size_t)bc * NSP + h * 24;
#pragma unroll
  for (int ch = 0; ch < 3; ++ch) {
    h8 st;
#pragma unroll
    for (int x = 0; x < 8; ++x) st[x] = (_Float16)o[ch * 8 + x];
    *(h8*)(op + ch * 8) = st;
  }
}

// ---------------- layout converters (tiled LDS transposes) ----------------
__global__ __launch_bounds__(256)
void k_nchw2t(const float* __restrict__ in, _Float16* __restrict__ out, int C)
{ // f32 [B][C][576] -> f16 [B*576][C]
  __shared__ float t[32][33];
  const int b = blockIdx.z, n0 = blockIdx.x * 32, c0 = blockIdx.y * 32;
  const int tx = threadIdx.x & 31, ty = threadIdx.x >> 5;
  for (int r = ty; r < 32; r += 8)
    t[r][tx] = in[((size_t)b * C + c0 + r) * NSP + n0 + tx];
  __syncthreads();
  for (int r = ty; r < 32; r += 8)
    out[((size_t)b * NSP + n0 + r) * C + c0 + tx] = (_Float16)t[tx][r];
}

__global__ __launch_bounds__(256)
void k_t2nchw(const _Float16* __restrict__ in, float* __restrict__ out, int C)
{ // f16 [B*576][C] -> f32 [B][C][576]
  __shared__ float t[32][33];
  const int b = blockIdx.z, n0 = blockIdx.x * 32, c0 = blockIdx.y * 32;
  const int tx = threadIdx.x & 31, ty = threadIdx.x >> 5;
  for (int r = ty; r < 32; r += 8)
    t[r][tx] = (float)in[((size_t)b * NSP + n0 + r) * C + c0 + tx];
  __syncthreads();
  for (int r = ty; r < 32; r += 8)
    out[((size_t)b * C + c0 + r) * NSP + n0 + tx] = t[tx][r];
}

__global__ __launch_bounds__(256)
void k_t2n16(const _Float16* __restrict__ in, _Float16* __restrict__ out, int C)
{ // f16 [B*576][C] -> f16 [B][C][576]
  __shared__ _Float16 t[32][40];
  const int b = blockIdx.z, n0 = blockIdx.x * 32, c0 = blockIdx.y * 32;
  const int tx = threadIdx.x & 31, ty = threadIdx.x >> 5;
  for (int r = ty; r < 32; r += 8)
    t[r][tx] = in[((size_t)b * NSP + n0 + r) * C + c0 + tx];
  __syncthreads();
  for (int r = ty; r < 32; r += 8)
    out[((size_t)b * C + c0 + r) * NSP + n0 + tx] = t[tx][r];
}

// ---------------- small prep kernels ----------------
__global__ void k_cvt(const float* __restrict__ in, _Float16* __restrict__ out, int n)
{
  const int i = blockIdx.x * 256 + threadIdx.x;
  if (i < n) out[i] = (_Float16)in[i];
}

__global__ void k_fuse(const float* __restrict__ bias, const float* __restrict__ bns,
                       const float* __restrict__ bnb, float* __restrict__ s,
                       float* __restrict__ b, int n)
{ // out = gemm*s + b, with BN folded: s=bn_s, b=bias*bn_s+bn_b (or s=1,b=bias)
  const int i = blockIdx.x * 256 + threadIdx.x;
  if (i >= n) return;
  const float sc = bns ? bns[i] : 1.f;
  const float bb = bnb ? bnb[i] : 0.f;
  s[i] = sc;
  b[i] = bias[i] * sc + bb;
}

// =====================================================================
// host orchestration
// =====================================================================
extern "C" void kernel_launch(void* const* d_in, const int* in_sizes, int n_in,
                              void* d_out, int out_size, void* d_ws, size_t ws_size,
                              hipStream_t stream)
{
  (void)in_sizes; (void)n_in; (void)out_size; (void)ws_size;
  auto F = [&](int i) { return (const float*)d_in[i]; };
  const float* inx = F(0);
  const float* iny = F(1);

  // ---- workspace arena ----
  char* base = (char*)d_ws;
  size_t off = 0;
  auto alloc = [&](size_t bytes) { void* r = base + off; off += (bytes + 255) & ~(size_t)255; return r; };
  auto ah  = [&](size_t n) { return (_Float16*)alloc(n * 2); };
  auto af32 = [&](size_t n) { return (float*)alloc(n * 4); };

  auto cvt = [&](const float* s, _Float16* d, int n) {
    k_cvt<<<dim3((n + 255) / 256), dim3(256), 0, stream>>>(s, d, n);
  };
  auto fuse = [&](const float* bias, const float* s, const float* b, float* os, float* ob, int n) {
    k_fuse<<<dim3((n + 255) / 256), dim3(256), 0, stream>>>(bias, s, b, os, ob, n);
  };
  auto gemm = [&](const _Float16* A, const _Float16* B, _Float16* O,
                  int M, int N, int K, int lda, int ldb, int ldo,
                  int ZH, int aZb, int aZh, int bZb, int bZh, int oZb, int oZh,
                  const float* sv, const float* bv, float ks, int act,
                  const _Float16* res, int nz) {
    dim3 g((M + 127) / 128, (N + 127) / 128, nz);
    k_wmma_gemm<<<g, dim3(128), 0, stream>>>(A, B, O, M, N, K, lda, ldb, ldo,
                                             ZH, aZb, aZh, bZb, bZh, oZb, oZh,
                                             sv, bv, ks, act, res);
  };

  // ---- weight/param prep ----
  struct AttW { _Float16 *wq, *wk, *wv, *wp; const float* dw;
                float *qs, *qb, *ks, *kb, *vs, *vb, *ls, *lb, *ps, *pb;
                const float *t1w, *t1b, *t2w, *t2b; };
  struct MlpW { _Float16 *w1, *w2; float *s1, *b1, *s2, *b2; };

  auto mkatt = [&](int bi) {
    AttW w;
    w.wq = ah(QKC * CIN);  cvt(F(bi + 0),  w.wq, QKC * CIN);
    w.wk = ah(QKC * CIN);  cvt(F(bi + 4),  w.wk, QKC * CIN);
    w.wv = ah(DHC * CIN);  cvt(F(bi + 8),  w.wv, DHC * CIN);
    w.wp = ah(CIN * DHC);  cvt(F(bi + 20), w.wp, CIN * DHC);
    w.dw = F(bi + 12);
    w.qs = af32(QKC); w.qb = af32(QKC); fuse(F(bi + 1),  F(bi + 2),  F(bi + 3),  w.qs, w.qb, QKC);
    w.ks = af32(QKC); w.kb = af32(QKC); fuse(F(bi + 5),  F(bi + 6),  F(bi + 7),  w.ks, w.kb, QKC);
    w.vs = af32(DHC); w.vb = af32(DHC); fuse(F(bi + 9),  F(bi + 10), F(bi + 11), w.vs, w.vb, DHC);
    w.ls = af32(DHC); w.lb = af32(DHC); fuse(F(bi + 13), F(bi + 14), F(bi + 15), w.ls, w.lb, DHC);
    w.ps = af32(CIN); w.pb = af32(CIN); fuse(F(bi + 21), F(bi + 22), F(bi + 23), w.ps, w.pb, CIN);
    w.t1w = F(bi + 16); w.t1b = F(bi + 17); w.t2w = F(bi + 18); w.t2b = F(bi + 19);
    return w;
  };
  auto mkmlp = [&](int bi) {
    MlpW m;
    m.w1 = ah(HIDC * CIN); cvt(F(bi + 0), m.w1, HIDC * CIN);
    m.w2 = ah(CIN * HIDC); cvt(F(bi + 4), m.w2, CIN * HIDC);
    m.s1 = af32(HIDC); m.b1 = af32(HIDC); fuse(F(bi + 1), F(bi + 2), F(bi + 3), m.s1, m.b1, HIDC);
    m.s2 = af32(CIN);  m.b2 = af32(CIN);  fuse(F(bi + 5), F(bi + 6), F(bi + 7), m.s2, m.b2, CIN);
    return m;
  };

  _Float16* wcx = ah(A3C * CIN); cvt(F(66), wcx, A3C * CIN);
  _Float16* wcy = ah(A3C * CIN); cvt(F(68), wcy, A3C * CIN);
  _Float16* wpx = ah(CIN * ADC); cvt(F(70), wpx, CIN * ADC);
  _Float16* wpy = ah(CIN * ADC); cvt(F(72), wpy, CIN * ADC);
  float *cxs = af32(A3C), *cxb = af32(A3C); fuse(F(67), nullptr, nullptr, cxs, cxb, A3C);
  float *cys = af32(A3C), *cyb = af32(A3C); fuse(F(69), nullptr, nullptr, cys, cyb, A3C);
  float *pxs = af32(CIN), *pxb = af32(CIN); fuse(F(71), nullptr, nullptr, pxs, pxb, CIN);
  float *pys = af32(CIN), *pyb = af32(CIN); fuse(F(73), nullptr, nullptr, pys, pyb, CIN);

  // ---- activation buffers ----
  _Float16* xT   = ah((size_t)BNR * CIN);
  _Float16* yT   = ah((size_t)BNR * CIN);
  _Float16* qT   = ah((size_t)BNR * QKC);
  _Float16* kT   = ah((size_t)BNR * QKC);
  _Float16* vT   = ah((size_t)BNR * DHC);
  _Float16* vN   = ah((size_t)BNR * DHC);
  _Float16* attb = ah((size_t)NB * NSP * NHD * NSP);
  _Float16* netT = ah((size_t)BNR * DHC);
  _Float16* hid  = ah((size_t)BNR * HIDC);
  _Float16* x1   = ah((size_t)BNR * CIN);
  _Float16* x2   = ah((size_t)BNR * CIN);
  _Float16* y1   = ah((size_t)BNR * CIN);
  _Float16* y2   = ah((size_t)BNR * CIN);
  _Float16* qkvx = ah((size_t)BNR * A3C);
  _Float16* qkvy = ah((size_t)BNR * A3C);
  _Float16* oxT  = ah((size_t)BNR * ADC);
  _Float16* oyT  = ah((size_t)BNR * ADC);
  _Float16* xo   = ah((size_t)BNR * CIN);
  _Float16* yo   = ah((size_t)BNR * CIN);

  // ---- inputs to T layout ----
  k_nchw2t<<<dim3(18, 4, NB), dim3(256), 0, stream>>>(inx, xT, CIN);
  k_nchw2t<<<dim3(18, 4, NB), dim3(256), 0, stream>>>(iny, yT, CIN);

  AttW ax = mkatt(2), ay = mkatt(26);
  MlpW mx = mkmlp(50), my = mkmlp(58), mx2 = mkmlp(74), my2 = mkmlp(82);

  auto run_att = [&](const AttW& w, const MlpW& m, _Float16* inT,
                     _Float16* o1, _Float16* o2) {
    gemm(w.wq, inT, qT, QKC, BNR, CIN, CIN, CIN, QKC, 1,0,0,0,0,0,0, w.qs, w.qb, 1.f, 0, nullptr, 1);
    gemm(w.wk, inT, kT, QKC, BNR, CIN, CIN, CIN, QKC, 1,0,0,0,0,0,0, w.ks, w.kb, 1.f, 0, nullptr, 1);
    gemm(w.wv, inT, vT, DHC, BNR, CIN, CIN, CIN, DHC, 1,0,0,0,0,0,0, w.vs, w.vb, 1.f, 0, nullptr, 1);
    k_t2n16<<<dim3(18, DHC / 32, NB), dim3(256), 0, stream>>>(vT, vN, DHC);          // -> NCHW
    k_dwconv<<<dim3((NB * DHC * 24 + 255) / 256), dim3(256), 0, stream>>>(vN, w.dw, w.ls, w.lb, vT);
    // att[b][n][h][m] = (q . k) * dk^-0.5 : A rows = k (m), B rows = q (n)
    gemm(kT, qT, attb, NSP, NSP, DKH, QKC, QKC, NHD * NSP,
         NHD, NSP * QKC, DKH, NSP * QKC, DKH, NSP * NHD * NSP, NSP,
         nullptr, nullptr, 0.125f, 0, nullptr, NB * NHD);
    k_softmax_th<<<dim3(BNR), dim3(32), 0, stream>>>(attb, w.t1w, w.t1b, w.t2w, w.t2b, 1);
    // net[b][n][h*256+d] = relu(att @ v) : A rows = v NCHW (d), B rows = att (n)
    gemm(vT, attb, netT, DVH, NSP, NSP, NSP, NHD * NSP, DHC,
         NHD, DHC * NSP, DVH * NSP, NSP * NHD * NSP, NSP, NSP * DHC, DVH,
         nullptr, nullptr, 1.f, 1, nullptr, NB * NHD);
    gemm(w.wp, netT, o1, CIN, BNR, DHC, DHC, DHC, CIN, 1,0,0,0,0,0,0, w.ps, w.pb, 1.f, 0, inT, 1);
    gemm(m.w1, o1, hid, HIDC, BNR, CIN, CIN, CIN, HIDC, 1,0,0,0,0,0,0, m.s1, m.b1, 1.f, 2, nullptr, 1);
    gemm(m.w2, hid, o2, CIN, BNR, HIDC, HIDC, HIDC, CIN, 1,0,0,0,0,0,0, m.s2, m.b2, 1.f, 0, o1, 1);
  };

  run_att(ax, mx, xT, x1, x2);
  run_att(ay, my, yT, y1, y2);

  // ---- cross attention ----
  gemm(wcx, x2, qkvx, A3C, BNR, CIN, CIN, CIN, A3C, 1,0,0,0,0,0,0, cxs, cxb, 1.f, 0, nullptr, 1);
  gemm(wcy, y2, qkvy, A3C, BNR, CIN, CIN, CIN, A3C, 1,0,0,0,0,0,0, cys, cyb, 1.f, 0, nullptr, 1);
  const float gsc = 0.35355339059327373f;  // 8^-0.5
  // ax = softmax(qx . ky^T * g) : A rows = ky (qkvy+64), B rows = qx, K=8 zero-padded
  gemm(qkvy + 64, qkvx, attb, NSP, NSP, 8, A3C, A3C, NHD * NSP,
       NHD, NSP * A3C, 8, NSP * A3C, 8, NSP * NHD * NSP, NSP,
       nullptr, nullptr, gsc, 0, nullptr, NB * NHD);
  k_softmax_th<<<dim3(BNR), dim3(32), 0, stream>>>(attb, nullptr, nullptr, nullptr, nullptr, 0);
  k_cross_av<<<dim3(72), dim3(256), 0, stream>>>(attb, qkvy, oxT);
  gemm(wpx, oxT, x1, CIN, BNR, ADC, ADC, ADC, CIN, 1,0,0,0,0,0,0, pxs, pxb, 1.f, 0, x2, 1);
  // ay side
  gemm(qkvx + 64, qkvy, attb, NSP, NSP, 8, A3C, A3C, NHD * NSP,
       NHD, NSP * A3C, 8, NSP * A3C, 8, NSP * NHD * NSP, NSP,
       nullptr, nullptr, gsc, 0, nullptr, NB * NHD);
  k_softmax_th<<<dim3(BNR), dim3(32), 0, stream>>>(attb, nullptr, nullptr, nullptr, nullptr, 0);
  k_cross_av<<<dim3(72), dim3(256), 0, stream>>>(attb, qkvx, oyT);
  gemm(wpy, oyT, y1, CIN, BNR, ADC, ADC, ADC, CIN, 1,0,0,0,0,0,0, pys, pyb, 1.f, 0, y2, 1);

  // ---- final UFFNs (no residual) ----
  gemm(mx2.w1, x1, hid, HIDC, BNR, CIN, CIN, CIN, HIDC, 1,0,0,0,0,0,0, mx2.s1, mx2.b1, 1.f, 2, nullptr, 1);
  gemm(mx2.w2, hid, xo,  CIN, BNR, HIDC, HIDC, HIDC, CIN, 1,0,0,0,0,0,0, mx2.s2, mx2.b2, 1.f, 0, nullptr, 1);
  gemm(my2.w1, y1, hid, HIDC, BNR, CIN, CIN, CIN, HIDC, 1,0,0,0,0,0,0, my2.s1, my2.b1, 1.f, 2, nullptr, 1);
  gemm(my2.w2, hid, yo,  CIN, BNR, HIDC, HIDC, HIDC, CIN, 1,0,0,0,0,0,0, my2.s2, my2.b2, 1.f, 0, nullptr, 1);

  float* outx = (float*)d_out;
  float* outy = outx + (size_t)NB * CIN * NSP;
  k_t2nchw<<<dim3(18, 4, NB), dim3(256), 0, stream>>>(xo, outx, CIN);
  k_t2nchw<<<dim3(18, 4, NB), dim3(256), 0, stream>>>(yo, outy, CIN);
}
